// DaRnn_21852793602803
// MI455X (gfx1250) — matrix operand order repaired and checked by reading, not compiled, not run on hardware
//
#include <hip/hip_runtime.h>
#include <hip/hip_bf16.h>

// DA-RNN forward for MI455X (gfx1250, wave32, WMMA).
// Key math identity: encoder input attention softmax(feat + base[:,None]) is
// shift-invariant in base -> attn is constant over time; precompute Xw = attn*X.
// All GEMMs run through v_wmma_f32_16x16x32_bf16 (f32 accumulate), with a
// ping-pong (unroll-by-2) K pipeline: fragment loads overlap the WMMA chain
// and no register copies are needed for buffer rotation.

#define B_  1024
#define T_  64
#define N_  128
#define HE_ 256

typedef __attribute__((ext_vector_type(16))) __bf16 v16bf;
typedef __attribute__((ext_vector_type(8)))  float  v8f;

union FragU { v16bf v; uint4 u[2]; };

__device__ __forceinline__ unsigned short f2bf(float f) {
  unsigned int u = __float_as_uint(f);
  u += 0x7FFFu + ((u >> 16) & 1u);          // round-to-nearest-even
  return (unsigned short)(u >> 16);
}
__device__ __forceinline__ float bf2f(unsigned short s) {
  return __uint_as_float(((unsigned int)s) << 16);
}
__device__ __forceinline__ float sigm(float x) { return 1.0f / (1.0f + __expf(-x)); }

// load an A-fragment (two 16B chunks, 16B apart) / B-fragment (two 16B chunks,
// 16B apart) -- contiguous k order (lane half h holds k = 16h..16h+15), same for A 16x32 / B 32x16.
__device__ __forceinline__ void load_fa(FragU& f, const unsigned short* p) {
  f.u[0] = *(const uint4*)(p);
  f.u[1] = *(const uint4*)(p + 8);
}
__device__ __forceinline__ void load_fb(FragU& f, const unsigned short* p) {
  f.u[0] = *(const uint4*)(p);
  f.u[1] = *(const uint4*)(p + 8);
}

// ---------------------------------------------------------------------------
// WMMA GEMM: C[M x N] = [A0 | A1] (bf16, row-major, split K = k0 + k1)
//            x Bw^T   (Bw bf16, row-major N x K, ldb)
// Block: 8 waves, each wave does a 16(M) x 64(N) strip; block tile 128 x 64.
// Grid: (N/64, M/128). k0,k1 multiples of 32; total K a multiple of 64 and
// >= 128 (true at every call site: 384, 256, 512, 256).
// Ping-pong pipelined: tile loads issue a full tile ahead of their WMMAs.
// ---------------------------------------------------------------------------
__global__ __launch_bounds__(256)
void wmma_gemm_bf16(const unsigned short* __restrict__ A0, int lda0, int k0,
                    const unsigned short* __restrict__ A1, int lda1, int k1,
                    const unsigned short* __restrict__ Bw, int ldb,
                    void* __restrict__ Cout, int ldc, int bf16out) {
  const int lane = threadIdx.x & 31;
  const int wave = threadIdx.x >> 5;
  const int n0   = blockIdx.x * 64;
  const int mw   = blockIdx.y * 128 + wave * 16;
  const int l15  = lane & 15;
  const int hi16 = lane >> 4;            // 0 or 1
  const int ka   = hi16 << 4;            // A half-offset: 0 or 16
  const int kb   = hi16 << 4;            // B half-offset: 0 or 16

  v8f acc[4];
#pragma unroll
  for (int i = 0; i < 4; ++i)
#pragma unroll
    for (int r = 0; r < 8; ++r) acc[i][r] = 0.0f;

  // per-row base pointers (segment-relative K offsets added in the loop)
  const unsigned short* arow0 = A0 + (size_t)(mw + l15) * lda0 + ka;
  const unsigned short* arow1 = A1 + (size_t)(mw + l15) * lda1 + ka;
  const unsigned short* brow[4];
#pragma unroll
  for (int nf = 0; nf < 4; ++nf)
    brow[nf] = Bw + (size_t)(n0 + nf * 16 + l15) * ldb + kb;

  const int K = k0 + k1;
  auto aptr = [&](int kt) {
    return (kt < k0) ? (arow0 + kt) : (arow1 + (kt - k0));
  };

  // prologue: tiles 0 and 32 into ping/pong buffers
  FragU aP, bP[4], aQ, bQ[4];
  load_fa(aP, aptr(0));
#pragma unroll
  for (int nf = 0; nf < 4; ++nf) load_fb(bP[nf], brow[nf]);
  load_fa(aQ, aptr(32));
#pragma unroll
  for (int nf = 0; nf < 4; ++nf) load_fb(bQ[nf], brow[nf] + 32);

  for (int kt = 0; kt < K; kt += 64) {
    // prefetch targets (clamped: last iteration harmlessly re-reads tail tiles)
    const int kp = (kt + 64 < K) ? (kt + 64) : (K - 64);
    const int kq = (kt + 96 < K) ? (kt + 96) : (K - 32);

    // consume ping (tile kt), refill with tile kp
#pragma unroll
    for (int nf = 0; nf < 4; ++nf)
      acc[nf] = __builtin_amdgcn_wmma_f32_16x16x32_bf16(
          false, aP.v, false, bP[nf].v, (short)0, acc[nf], false, false);
    load_fa(aP, aptr(kp));
#pragma unroll
    for (int nf = 0; nf < 4; ++nf) load_fb(bP[nf], brow[nf] + kp);

    // consume pong (tile kt+32), refill with tile kq
#pragma unroll
    for (int nf = 0; nf < 4; ++nf)
      acc[nf] = __builtin_amdgcn_wmma_f32_16x16x32_bf16(
          false, aQ.v, false, bQ[nf].v, (short)0, acc[nf], false, false);
    load_fa(aQ, aptr(kq));
#pragma unroll
    for (int nf = 0; nf < 4; ++nf) load_fb(bQ[nf], brow[nf] + kq);
  }

  const int rbase = mw + (hi16 << 3);
#pragma unroll
  for (int nf = 0; nf < 4; ++nf) {
    const int col = n0 + nf * 16 + l15;
#pragma unroll
    for (int r = 0; r < 8; ++r) {
      size_t idx = (size_t)(rbase + r) * ldc + col;
      if (bf16out) ((unsigned short*)Cout)[idx] = f2bf(acc[nf][r]);
      else         ((float*)Cout)[idx] = acc[nf][r];
    }
  }
}

// ---------------------------------------------------------------------------
// Strided f32 -> bf16 convert/copy (weight packing)
// ---------------------------------------------------------------------------
__global__ void cvt_f32_bf16(const float* __restrict__ src, int sld,
                             unsigned short* __restrict__ dst, int dld,
                             int cols, int total) {
  int i = blockIdx.x * blockDim.x + threadIdx.x;
  if (i >= total) return;
  int r = i / cols, cc = i - r * cols;
  dst[(size_t)r * dld + cc] = f2bf(src[(size_t)r * sld + cc]);
}

// ---------------------------------------------------------------------------
// feat_scores -> softmax (time-invariant!) -> Xw = attn * X  (bf16)
// one block per batch row b, 128 threads (one per feature n)
// ---------------------------------------------------------------------------
__global__ __launch_bounds__(128)
void feat_attn_xw(const float* __restrict__ X, const float* __restrict__ Wa,
                  unsigned short* __restrict__ Xw) {
  __shared__ float red[128];
  __shared__ float smax, ssum;
  const int b = blockIdx.x, n = threadIdx.x;
  const float* xb = X + (size_t)b * T_ * N_;
  float acc = 0.f;
  for (int t = 0; t < T_; ++t) acc += xb[t * N_ + n] * Wa[2 * HE_ + t];
  red[n] = acc; __syncthreads();
  for (int s = 64; s > 0; s >>= 1) { if (n < s) red[n] = fmaxf(red[n], red[n + s]); __syncthreads(); }
  if (n == 0) smax = red[0];
  __syncthreads();
  float e = __expf(acc - smax);
  red[n] = e; __syncthreads();
  for (int s = 64; s > 0; s >>= 1) { if (n < s) red[n] += red[n + s]; __syncthreads(); }
  if (n == 0) ssum = red[0];
  __syncthreads();
  const float attn = e / ssum;
  unsigned short* xwb = Xw + (size_t)b * T_ * N_;
  for (int t = 0; t < T_; ++t) xwb[t * N_ + n] = f2bf(attn * xb[t * N_ + n]);
}

// ---------------------------------------------------------------------------
// Encoder LSTM cell: gates(B x 1024 f32) -> h,c ; emits h in bf16 (GEMM input
// for next step) and into input_encoded (bf16).  block=b, thread=j (256)
// ---------------------------------------------------------------------------
__global__ __launch_bounds__(256)
void enc_cell(const float* __restrict__ gates, const float* __restrict__ be_ih,
              const float* __restrict__ be_hh, float* __restrict__ c,
              unsigned short* __restrict__ h_bf, unsigned short* __restrict__ enc_bf,
              int t) {
  const int b = blockIdx.x, j = threadIdx.x;
  const size_t gb = (size_t)b * 1024;
  float gi = gates[gb + j]       + be_ih[j]       + be_hh[j];
  float gf = gates[gb + 256 + j] + be_ih[256 + j] + be_hh[256 + j];
  float gg = gates[gb + 512 + j] + be_ih[512 + j] + be_hh[512 + j];
  float go = gates[gb + 768 + j] + be_ih[768 + j] + be_hh[768 + j];
  const size_t hi = (size_t)b * 256 + j;
  float c2 = sigm(gf) * c[hi] + sigm(gi) * tanhf(gg);
  float hn = sigm(go) * tanhf(c2);
  c[hi] = c2;
  unsigned short hb = f2bf(hn);
  h_bf[hi] = hb;
  enc_bf[((size_t)b * T_ + t) * 256 + j] = hb;
}

// ---------------------------------------------------------------------------
// Decoder attention step (one block per b, 256 threads):
//   score[t2] = sum_k tanh(enc_proj[b,t2,k] + b1[k] + dproj[b,k]) * W2[k] + b2
//   attn = softmax(score) ; context[k] = sum_t2 attn[t2]*enc[b,t2,k]
//   y_tilde = <Wfc[0:256], context> + Wfc[256]*y_t + bfc
// ---------------------------------------------------------------------------
__global__ __launch_bounds__(256)
void dec_attn(const unsigned short* __restrict__ encproj,
              const unsigned short* __restrict__ encbf,
              const float* __restrict__ dproj, const float* __restrict__ b1,
              const float* __restrict__ W2, const float* __restrict__ b2,
              const float* __restrict__ Wfc, const float* __restrict__ bfc,
              const float* __restrict__ yhist, float* __restrict__ ctx,
              float* __restrict__ ytilde, int t) {
  __shared__ float sdp[256], sw2[256], ssc[64], sattn[64], sred[64];
  __shared__ float smax, ssum;
  const int b = blockIdx.x, tid = threadIdx.x;
  sdp[tid] = dproj[(size_t)b * 256 + tid] + b1[tid];
  sw2[tid] = W2[tid];
  __syncthreads();
  // scores: 64 groups of 4 lanes
  const int g = tid >> 2, l = tid & 3;
  const unsigned short* ep = encproj + ((size_t)b * T_ + g) * 256;
  float p = 0.f;
  for (int k = l; k < 256; k += 4) p += tanhf(bf2f(ep[k]) + sdp[k]) * sw2[k];
  p += __shfl_xor(p, 1, 32);
  p += __shfl_xor(p, 2, 32);
  if (l == 0) ssc[g] = p + b2[0];
  __syncthreads();
  // softmax over 64
  if (tid < 64) sred[tid] = ssc[tid];
  __syncthreads();
  for (int s = 32; s > 0; s >>= 1) { if (tid < s) sred[tid] = fmaxf(sred[tid], sred[tid + s]); __syncthreads(); }
  if (tid == 0) smax = sred[0];
  __syncthreads();
  if (tid < 64) { float e = __expf(ssc[tid] - smax); sattn[tid] = e; sred[tid] = e; }
  __syncthreads();
  for (int s = 32; s > 0; s >>= 1) { if (tid < s) sred[tid] += sred[tid + s]; __syncthreads(); }
  if (tid == 0) ssum = sred[0];
  __syncthreads();
  const float inv = 1.f / ssum;
  // context (thread = k)
  const unsigned short* eb = encbf + (size_t)b * T_ * 256 + tid;
  float cx = 0.f;
  for (int tt = 0; tt < T_; ++tt) cx += (sattn[tt] * inv) * bf2f(eb[(size_t)tt * 256]);
  ctx[(size_t)b * 256 + tid] = cx;
  __syncthreads();
  sdp[tid] = Wfc[tid] * cx;
  __syncthreads();
  for (int s = 128; s > 0; s >>= 1) { if (tid < s) sdp[tid] += sdp[tid + s]; __syncthreads(); }
  if (tid == 0)
    ytilde[b] = sdp[0] + Wfc[256] * yhist[(size_t)b * T_ + t] + bfc[0];
}

// ---------------------------------------------------------------------------
// Decoder LSTM cell: gates(h part from GEMM) + y_tilde*Wd_ih (rank-1) + biases
// ---------------------------------------------------------------------------
__global__ __launch_bounds__(256)
void dec_cell(const float* __restrict__ gates, const float* __restrict__ Wd_ih,
              const float* __restrict__ bd_ih, const float* __restrict__ bd_hh,
              const float* __restrict__ ytilde, float* __restrict__ h,
              float* __restrict__ c, unsigned short* __restrict__ hbf,
              unsigned short* __restrict__ cbf) {
  const int b = blockIdx.x, j = threadIdx.x;
  const size_t gb = (size_t)b * 1024;
  const float yt = ytilde[b];
  float gi = gates[gb + j]       + yt * Wd_ih[j]       + bd_ih[j]       + bd_hh[j];
  float gf = gates[gb + 256 + j] + yt * Wd_ih[256 + j] + bd_ih[256 + j] + bd_hh[256 + j];
  float gg = gates[gb + 512 + j] + yt * Wd_ih[512 + j] + bd_ih[512 + j] + bd_hh[512 + j];
  float go = gates[gb + 768 + j] + yt * Wd_ih[768 + j] + bd_ih[768 + j] + bd_hh[768 + j];
  const size_t hi = (size_t)b * 256 + j;
  float c2 = sigm(gf) * c[hi] + sigm(gi) * tanhf(gg);
  float hn = sigm(go) * tanhf(c2);
  c[hi] = c2; h[hi] = hn;
  hbf[hi] = f2bf(hn);
  cbf[hi] = f2bf(c2);
}

// ---------------------------------------------------------------------------
// out[b] = <Wff[0:256], h[b]> + <Wff[256:512], context[b]> + bff
// ---------------------------------------------------------------------------
__global__ __launch_bounds__(256)
void final_out(const float* __restrict__ h, const float* __restrict__ ctx,
               const float* __restrict__ Wff, const float* __restrict__ bff,
               float* __restrict__ out) {
  __shared__ float red[256];
  const int b = blockIdx.x, j = threadIdx.x;
  red[j] = Wff[j] * h[(size_t)b * 256 + j] + Wff[256 + j] * ctx[(size_t)b * 256 + j];
  __syncthreads();
  for (int s = 128; s > 0; s >>= 1) { if (j < s) red[j] += red[j + s]; __syncthreads(); }
  if (j == 0) out[b] = red[0] + bff[0];
}

// ---------------------------------------------------------------------------
extern "C" void kernel_launch(void* const* d_in, const int* in_sizes, int n_in,
                              void* d_out, int out_size, void* d_ws, size_t ws_size,
                              hipStream_t stream) {
  const float* X      = (const float*)d_in[0];
  const float* y_hist = (const float*)d_in[1];
  const float* We_ih  = (const float*)d_in[2];
  const float* We_hh  = (const float*)d_in[3];
  const float* be_ih  = (const float*)d_in[4];
  const float* be_hh  = (const float*)d_in[5];
  const float* Wa     = (const float*)d_in[6];
  // d_in[7] = ba : unused (softmax shift-invariance)
  const float* W1     = (const float*)d_in[8];
  const float* b1     = (const float*)d_in[9];
  const float* W2     = (const float*)d_in[10];
  const float* b2     = (const float*)d_in[11];
  const float* Wd_ih  = (const float*)d_in[12];
  const float* Wd_hh  = (const float*)d_in[13];
  const float* bd_ih  = (const float*)d_in[14];
  const float* bd_hh  = (const float*)d_in[15];
  const float* Wfc    = (const float*)d_in[16];
  const float* bfc    = (const float*)d_in[17];
  const float* Wff    = (const float*)d_in[18];
  const float* bff    = (const float*)d_in[19];
  float* out = (float*)d_out;

  // ---- workspace carve-out (~97 MB) ----
  char* ws = (char*)d_ws;
  size_t off = 0;
  auto alloc = [&](size_t bytes) -> void* {
    void* p = ws + off;
    off = (off + bytes + 255) & ~(size_t)255;
    return p;
  };
  float*          gates  = (float*)alloc((size_t)B_ * 1024 * 4);
  float*          cenc   = (float*)alloc((size_t)B_ * HE_ * 4);
  unsigned short* hencbf = (unsigned short*)alloc((size_t)B_ * HE_ * 2);
  unsigned short* Xw     = (unsigned short*)alloc((size_t)B_ * T_ * N_ * 2);
  unsigned short* encbf  = (unsigned short*)alloc((size_t)B_ * T_ * HE_ * 2);
  unsigned short* eprojb = (unsigned short*)alloc((size_t)B_ * T_ * HE_ * 2);
  unsigned short* Wenc   = (unsigned short*)alloc((size_t)1024 * 384 * 2);
  unsigned short* W1hc   = (unsigned short*)alloc((size_t)256 * 512 * 2);
  unsigned short* W1e    = (unsigned short*)alloc((size_t)256 * 256 * 2);
  unsigned short* Wdhh   = (unsigned short*)alloc((size_t)1024 * 256 * 2);
  float*          dproj  = (float*)alloc((size_t)B_ * HE_ * 4);
  float*          hdec   = (float*)alloc((size_t)B_ * HE_ * 4);
  float*          cdec   = (float*)alloc((size_t)B_ * HE_ * 4);
  unsigned short* hdecbf = (unsigned short*)alloc((size_t)B_ * HE_ * 2);
  unsigned short* cdecbf = (unsigned short*)alloc((size_t)B_ * HE_ * 2);
  float*          ctx    = (float*)alloc((size_t)B_ * HE_ * 4);
  float*          yt     = (float*)alloc((size_t)B_ * 4);

  // ---- zero recurrent state (capture-safe memsets) ----
  hipMemsetAsync(cenc,   0, (size_t)B_ * HE_ * 4, stream);
  hipMemsetAsync(hencbf, 0, (size_t)B_ * HE_ * 2, stream);
  hipMemsetAsync(cdec,   0, (size_t)B_ * HE_ * 4, stream);
  hipMemsetAsync(hdecbf, 0, (size_t)B_ * HE_ * 2, stream);
  hipMemsetAsync(cdecbf, 0, (size_t)B_ * HE_ * 2, stream);

  // ---- pack weights to bf16 ----
  // Wenc = [We_ih | We_hh]  (1024 x 384)
  cvt_f32_bf16<<<(1024 * 128 + 255) / 256, 256, 0, stream>>>(We_ih, 128, Wenc, 384, 128, 1024 * 128);
  cvt_f32_bf16<<<(1024 * 256 + 255) / 256, 256, 0, stream>>>(We_hh, 256, Wenc + 128, 384, 256, 1024 * 256);
  // W1hc = W1[:, 0:512] (= [W1_h | W1_c]), W1e = W1[:, 512:768]
  cvt_f32_bf16<<<(256 * 512 + 255) / 256, 256, 0, stream>>>(W1, 768, W1hc, 512, 512, 256 * 512);
  cvt_f32_bf16<<<(256 * 256 + 255) / 256, 256, 0, stream>>>(W1 + 512, 768, W1e, 256, 256, 256 * 256);
  cvt_f32_bf16<<<(1024 * 256 + 255) / 256, 256, 0, stream>>>(Wd_hh, 256, Wdhh, 256, 256, 1024 * 256);

  // ---- time-invariant encoder attention: Xw = softmax(feat_scores) * X ----
  feat_attn_xw<<<B_, 128, 0, stream>>>(X, Wa, Xw);

  // ---- encoder: 64 sequential steps ----
  for (int t = 0; t < T_; ++t) {
    // gates = [Xw_t | h] @ [We_ih | We_hh]^T   (M=1024, N=1024, K=128+256)
    wmma_gemm_bf16<<<dim3(16, 8), 256, 0, stream>>>(
        Xw + t * N_, T_ * N_, 128, hencbf, 256, 256, Wenc, 384, gates, 1024, 0);
    enc_cell<<<B_, 256, 0, stream>>>(gates, be_ih, be_hh, cenc, hencbf, encbf, t);
  }

  // ---- enc_proj = input_encoded @ W1_e^T  (M=65536, N=256, K=256) -> bf16 ----
  wmma_gemm_bf16<<<dim3(4, 512), 256, 0, stream>>>(
      encbf, 256, 256, encbf, 256, 0, W1e, 256, eprojb, 256, 1);

  // ---- decoder: 64 sequential steps ----
  for (int t = 0; t < T_; ++t) {
    // dproj = [h | c] @ [W1_h | W1_c]^T   (M=1024, N=256, K=512)
    wmma_gemm_bf16<<<dim3(4, 8), 256, 0, stream>>>(
        hdecbf, 256, 256, cdecbf, 256, 256, W1hc, 512, dproj, 256, 0);
    dec_attn<<<B_, 256, 0, stream>>>(eprojb, encbf, dproj, b1, W2, b2,
                                     Wfc, bfc, y_hist, ctx, yt, t);
    // gates = h @ Wd_hh^T   (M=1024, N=1024, K=256)
    wmma_gemm_bf16<<<dim3(16, 8), 256, 0, stream>>>(
        hdecbf, 256, 256, hdecbf, 256, 0, Wdhh, 256, gates, 1024, 0);
    dec_cell<<<B_, 256, 0, stream>>>(gates, Wd_ih, bd_ih, bd_hh, yt,
                                     hdec, cdec, hdecbf, cdecbf);
  }

  // ---- output ----
  final_out<<<B_, 256, 0, stream>>>(hdec, ctx, Wff, bff, out);
}
